// RegionProposalNetwork_3882650436217
// MI455X (gfx1250) — matrix operand order repaired and checked by reading, not compiled, not run on hardware
//
#include <hip/hip_runtime.h>
#include <hip/hip_bf16.h>
#include <stdint.h>
#include <stddef.h>

typedef __attribute__((ext_vector_type(16))) __bf16 v16bf;
typedef __attribute__((ext_vector_type(8)))  float  v8f;

#define B_    4
#define CIN   1024
#define H_    50
#define W_    76
#define CO    512
#define KTOT  9216           // CIN * 9
#define HW    (H_ * W_)      // 3800
#define MHEAD 54
#define OBJ_TOTAL (B_ * HW * 18)

// pixel tile per block
#define ROWS 4
#define COLS 16
#define XT   5               // ceil(76/16)
#define YT   13              // ceil(50/4)
#define BANDROWS (ROWS + 2)  // 6
#define BANDCOLS (COLS + 2)  // 18

struct alignas(16) U4 { unsigned x, y, z, w; };

union Frag {
    U4       q[2];
    v16bf    bf;
    unsigned u[8];
};

__device__ __host__ __forceinline__ unsigned pack_bf16(float lo, float hi) {
    unsigned a = __builtin_bit_cast(unsigned, lo);
    unsigned b = __builtin_bit_cast(unsigned, hi);
    a = (a + 0x7fffu + ((a >> 16) & 1u)) >> 16;           // RNE to bf16, low half
    b = (b + 0x7fffu + ((b >> 16) & 1u)) & 0xffff0000u;   // RNE to bf16, high half
    return b | a;
}

// ---------------------------------------------------------------------------
// Repack conv weights: OIHW fp32 [512,1024,3,3] -> bf16 row-major A[m][k],
// k = (kh*3+kw)*1024 + ci, stored as packed bf16 pairs (one dword = 2 k's).
// ---------------------------------------------------------------------------
__global__ __launch_bounds__(256)
void repack_wfeat(const float* __restrict__ w, unsigned* __restrict__ dst) {
    int idx = blockIdx.x * 256 + threadIdx.x;     // 512 * 4608 dwords
    int m  = idx / 4608;
    int j  = idx - m * 4608;
    int k0 = j * 2;
    int p  = k0 >> 10;                            // filter tap 0..8
    int ci = k0 & 1023;                           // even
    const float* s = w + ((size_t)(m * CIN + ci)) * 9 + p;
    dst[idx] = pack_bf16(s[0], s[9]);             // ci, ci+1 (stride 9 in OIHW)
}

// ---------------------------------------------------------------------------
// Repack head weights: [18,512] + [36,512] fp32 -> bf16 A[64][512] (rows 54..63 = 0)
// ---------------------------------------------------------------------------
__global__ __launch_bounds__(256)
void repack_whead(const float* __restrict__ wobj, const float* __restrict__ wtr,
                  unsigned* __restrict__ dst) {
    int idx = blockIdx.x * 256 + threadIdx.x;     // 64 * 256 dwords
    int m  = idx >> 8;
    int c0 = (idx & 255) * 2;
    float lo = 0.f, hi = 0.f;
    if (m < 18)        { lo = wobj[m * 512 + c0];        hi = wobj[m * 512 + c0 + 1]; }
    else if (m < 54)   { lo = wtr[(m - 18) * 512 + c0];  hi = wtr[(m - 18) * 512 + c0 + 1]; }
    dst[idx] = pack_bf16(lo, hi);
}

// ---------------------------------------------------------------------------
// Fused conv3x3+ReLU (implicit GEMM, bf16 WMMA) + 1x1 heads + permuted store.
// Block: 256 threads = 8 waves; tile = 4 rows x 16 cols of pixels, M = 512.
// ---------------------------------------------------------------------------
__global__ __launch_bounds__(256)
void rpn_fused(const float* __restrict__ feat,
               const __bf16* __restrict__ wsA,   // [512][9216] bf16
               const __bf16* __restrict__ wsH,   // [64][512] bf16
               const float* __restrict__ bfeat,
               const float* __restrict__ bobj,
               const float* __restrict__ btr,
               float* __restrict__ out) {
    extern __shared__ char smem[];
    unsigned* hB    = (unsigned*)smem;                              // 16*64*16 dwords = 64 KB
    float*    sbias = (float*)(smem + 16384 * 4);                   // 512 floats
    float*    hbias = sbias + 512;                                  // 64 floats
    unsigned* band  = (unsigned*)(smem + 16384 * 4 + (512 + 64) * 4); // 1728 dwords

    const int tid   = threadIdx.x;
    const int wave  = tid >> 5;
    const int lane  = tid & 31;
    const int lanex = lane & 15;
    const int hi8   = lane >> 4;      // 0: lanes 0-15, 1: lanes 16-31

    const int gx = blockIdx.x;
    const int xt = gx % XT;
    const int yt = (gx / XT) % YT;
    const int bb = gx / (XT * YT);
    const int x0 = xt * COLS;
    const int y0 = yt * ROWS;

    // stage biases once
    for (int i = tid; i < CO; i += 256) sbias[i] = bfeat[i];
    if (tid < 64) hbias[tid] = (tid < 18) ? bobj[tid] : (tid < MHEAD ? btr[tid - 18] : 0.f);

    const v8f vzero = {0.f, 0.f, 0.f, 0.f, 0.f, 0.f, 0.f, 0.f};
    v8f acc[4][4];
#pragma unroll
    for (int i = 0; i < 4; ++i)
#pragma unroll
        for (int j = 0; j < 4; ++j) acc[i][j] = vzero;

    const float* featB = feat + (size_t)bb * CIN * HW;

    // ---- conv implicit GEMM: K = 9 taps x 1024 ci, chunks of 32 ci ----
    for (int cb = 0; cb < CIN; cb += 32) {
        __syncthreads();  // previous chunk's band reads complete
        // stage 6x18 band of 32 channels as packed bf16 pairs: band[row][col][c2]
        for (int idx = tid; idx < BANDROWS * BANDCOLS * 16; idx += 256) {
            int c2  = idx & 15;
            int col = (idx >> 4) % BANDCOLS;
            int row = idx / (16 * BANDCOLS);
            int yy = y0 + row - 1;
            int xx = x0 + col - 1;
            float lo = 0.f, hi = 0.f;
            if (yy >= 0 && yy < H_ && xx >= 0 && xx < W_) {
                const float* p = featB + (size_t)(cb + 2 * c2) * HW + yy * W_ + xx;
                lo = p[0];
                hi = p[HW];
            }
            band[(row * BANDCOLS + col) * 16 + c2] = pack_bf16(lo, hi);
        }
        __syncthreads();

        // prefetch next chunk of the (L2-resident) repacked A stream
        if (cb + 32 < CIN)
            __builtin_prefetch(wsA + (size_t)(wave * 64 + lanex) * KTOT + cb + 32, 0, 1);

        for (int p = 0; p < 9; ++p) {
            const int kh = p / 3, kw = p - kh * 3;
            // B fragments (32x16 bf16) per N-tile: two ds_load_b128 (K{0-7,16-23} / K{8-15,24-31})
            Frag bfr[4];
#pragma unroll
            for (int nt = 0; nt < 4; ++nt) {
                const U4* bp = (const U4*)(band + (((nt + kh) * BANDCOLS + lanex + kw) * 16 + hi8 * 4));
                bfr[nt].q[0] = bp[0];
                bfr[nt].q[1] = bp[2];
            }
            const int kg = p * CIN + cb;
#pragma unroll
            for (int mt = 0; mt < 4; ++mt) {
                const int m = (wave * 4 + mt) * 16 + lanex;
                // A fragment (16x32 bf16): lanes 0-15 K{0-7,16-23}, lanes 16-31 K{8-15,24-31}
                const __bf16* ap = wsA + (size_t)m * KTOT + kg + hi8 * 8;
                Frag af;
                af.q[0] = *(const U4*)(ap);
                af.q[1] = *(const U4*)(ap + 16);
#pragma unroll
                for (int nt = 0; nt < 4; ++nt)
                    acc[mt][nt] = __builtin_amdgcn_wmma_f32_16x16x32_bf16(
                        false, af.bf, false, bfr[nt].bf, (short)0, acc[mt][nt], false, false);
            }
        }
    }
    __syncthreads();

    // ---- epilogue: bias + ReLU, repack h into LDS as head-GEMM B fragments ----
#pragma unroll
    for (int mt = 0; mt < 4; ++mt) {
        const int mbase = (wave * 4 + mt) * 16 + hi8 * 8;
#pragma unroll
        for (int nt = 0; nt < 4; ++nt) {
            v8f a = acc[mt][nt];
#pragma unroll
            for (int r = 0; r < 8; ++r) {
                float v = a[r] + sbias[mbase + r];
                a[r] = v > 0.f ? v : 0.f;
            }
            const int n = nt * 16 + lanex;
#pragma unroll
            for (int pa = 0; pa < 4; ++pa) {
                const int c2g = (wave * 4 + mt) * 8 + hi8 * 4 + pa;  // channel pair index
                const int s   = c2g >> 4;
                const int c2l = c2g & 15;
                hB[(s * 64 + n) * 16 + c2l] = pack_bf16(a[2 * pa], a[2 * pa + 1]);
            }
        }
    }
    __syncthreads();

    // ---- head GEMM: out[64][64] = Whead[64][512] x h[512][64], waves 0-3 ----
    if (wave < 4) {
        v8f hacc[4];
#pragma unroll
        for (int nt = 0; nt < 4; ++nt) hacc[nt] = vzero;

        for (int s = 0; s < 16; ++s) {
            Frag af;
            const __bf16* ap = wsH + (size_t)(wave * 16 + lanex) * 512 + s * 32 + hi8 * 8;
            af.q[0] = *(const U4*)(ap);
            af.q[1] = *(const U4*)(ap + 16);
#pragma unroll
            for (int nt = 0; nt < 4; ++nt) {
                Frag bfh;
                const U4* bp = (const U4*)(hB + ((s * 64 + nt * 16 + lanex) * 16 + hi8 * 4));
                bfh.q[0] = bp[0];
                bfh.q[1] = bp[2];
                hacc[nt] = __builtin_amdgcn_wmma_f32_16x16x32_bf16(
                    false, af.bf, false, bfh.bf, (short)0, hacc[nt], false, false);
            }
        }

        float* outObj = out;
        float* outTr  = out + OBJ_TOTAL;
#pragma unroll
        for (int nt = 0; nt < 4; ++nt) {
            const int y = y0 + nt;
            const int x = x0 + lanex;
            if (y < H_ && x < W_) {
                const int pix = y * W_ + x;
#pragma unroll
                for (int r = 0; r < 8; ++r) {
                    const int mh = wave * 16 + hi8 * 8 + r;
                    const float v = hacc[nt][r] + hbias[mh];
                    if (mh < 18)
                        outObj[(size_t)bb * HW * 18 + (size_t)pix * 18 + mh] = v;
                    else if (mh < MHEAD)
                        outTr[(size_t)bb * HW * 36 + (size_t)pix * 36 + (mh - 18)] = v;
                }
            }
        }
    }
}

// ---------------------------------------------------------------------------
extern "C" void kernel_launch(void* const* d_in, const int* in_sizes, int n_in,
                              void* d_out, int out_size, void* d_ws, size_t ws_size,
                              hipStream_t stream) {
    (void)in_sizes; (void)n_in; (void)out_size; (void)ws_size;
    const float* feat  = (const float*)d_in[0];
    const float* wfeat = (const float*)d_in[1];
    const float* bfeat = (const float*)d_in[2];
    const float* wobj  = (const float*)d_in[3];
    const float* bobj  = (const float*)d_in[4];
    const float* wtr   = (const float*)d_in[5];
    const float* btr   = (const float*)d_in[6];
    float* out = (float*)d_out;

    unsigned* wsA32 = (unsigned*)d_ws;            // 512*4608 dwords = 9.44 MB
    unsigned* wsH32 = wsA32 + 512 * 4608;         // 16384 dwords   = 64 KB
    const __bf16* wsA = (const __bf16*)wsA32;
    const __bf16* wsH = (const __bf16*)wsH32;

    repack_wfeat<<<(512 * 4608) / 256, 256, 0, stream>>>(wfeat, wsA32);
    repack_whead<<<64, 256, 0, stream>>>(wobj, wtr, wsH32);

    const size_t smem = 16384 * 4 + (512 + 64) * 4 + 1728 * 4;  // 74752 bytes
    rpn_fused<<<dim3(B_ * YT * XT), 256, smem, stream>>>(feat, wsA, wsH, bfeat, bobj, btr, out);
}
